// GAT_4312147165894
// MI455X (gfx1250) — matrix operand order repaired and checked by reading, not compiled, not run on hardware
//
#include <hip/hip_runtime.h>
#include <hip/hip_bf16.h>

typedef __bf16 bf16_t;
typedef __attribute__((ext_vector_type(16))) __bf16 v16bf;
typedef __attribute__((ext_vector_type(8)))  float  v8f;

#define N_NODES   20000
#define E_EDGES   640000
#define E_TOT     (E_EDGES + N_NODES)   /* 660000: edges + self loops */
#define F_IN      128
#define HID       32
#define HEADS     8
#define D1        (HID * HEADS)         /* 256 */
#define NCLS      40
#define NCLS_PAD  48                    /* padded to 3 WMMA column tiles */
#define NEG_SLOPE 0.2f
#define ENC_NEG_INF 0x007FFFFFu         /* order-preserving encoding of -inf */

// ---------------------------------------------------------------- helpers
__device__ __forceinline__ unsigned encf(float f) {
    unsigned u = __float_as_uint(f);
    return (u & 0x80000000u) ? ~u : (u | 0x80000000u);
}
__device__ __forceinline__ float decf(unsigned u) {
    return __uint_as_float((u & 0x80000000u) ? (u & 0x7FFFFFFFu) : ~u);
}
__device__ __forceinline__ void edge_nodes(const long long* ei, int e, int& s, int& d) {
    if (e < E_EDGES) { s = (int)ei[e]; d = (int)ei[E_EDGES + e]; }
    else             { s = d = e - E_EDGES; }   // appended self-loops
}

// ---------------------------------------------------------------- fills
__global__ void fill_u32(unsigned* p, unsigned val, int n) {
    int i = blockIdx.x * blockDim.x + threadIdx.x;
    if (i < n) p[i] = val;
}

// ---------------------------------------------------------------- fp32 -> bf16
__global__ void f2bf_kernel(const float* __restrict__ in, bf16_t* __restrict__ out, int n) {
    int i = blockIdx.x * blockDim.x + threadIdx.x;
    if (i < n) out[i] = (bf16_t)in[i];
}

// transpose+convert: W[rows=K][cols] -> Wt[colsPad][rows], zero-padded columns
__global__ void transpose_bf_kernel(const float* __restrict__ W, bf16_t* __restrict__ Wt,
                                    int rows, int cols, int colsPad) {
    int i = blockIdx.x * blockDim.x + threadIdx.x;
    if (i >= rows * colsPad) return;
    int c = i / rows, k = i - c * rows;
    Wt[i] = (c < cols) ? (bf16_t)W[(size_t)k * cols + c] : (bf16_t)0.f;
}

// ---------------------------------------------------------------- WMMA GEMM
// C[M x (16*nColTiles)] = A[M x K] * Bt^T, A row-major bf16, Bt[ncols][K] bf16
__global__ void gemm_bf16_wmma(const bf16_t* __restrict__ A, const bf16_t* __restrict__ Bt,
                               float* __restrict__ C, int K, int nColTiles, int ldc,
                               int totalTiles) {
    int gwave = (blockIdx.x * blockDim.x + threadIdx.x) >> 5;   // wave-uniform
    if (gwave >= totalTiles) return;                            // uniform exit
    int lane = threadIdx.x & 31;
    int rowTile = gwave / nColTiles;
    int colTile = gwave - rowTile * nColTiles;
    int half = lane >> 4;       // 0: lanes 0-15, 1: lanes 16-31
    int mn   = lane & 15;       // A-row / B-column owned by this lane

    const bf16_t* arow = A  + (size_t)(rowTile * 16 + mn) * K;
    const bf16_t* brow = Bt + (size_t)(colTile * 16 + mn) * K;

    v8f acc = {0.f, 0.f, 0.f, 0.f, 0.f, 0.f, 0.f, 0.f};
    for (int k0 = 0; k0 < K; k0 += 32) {
        v16bf a, b;
#pragma unroll
        for (int j = 0; j < 8; ++j) {
            // A pair j: K = {0|16} + (j&3)*2 (+8 for high half-wave)
            int ka = k0 + ((j < 4) ? 0 : 16) + (j & 3) * 2 + half * 8;
            a[2 * j]     = arow[ka];
            a[2 * j + 1] = arow[ka + 1];
            // B pair j: same K order as A
            int kb = k0 + ((j < 4) ? 0 : 16) + (j & 3) * 2 + half * 8;
            b[2 * j]     = brow[kb];
            b[2 * j + 1] = brow[kb + 1];
        }
        acc = __builtin_amdgcn_wmma_f32_16x16x32_bf16(false, a, false, b,
                                                      (short)0, acc, false, false);
    }
#pragma unroll
    for (int r = 0; r < 8; ++r) {
        int row = rowTile * 16 + half * 8 + r;   // C layout: VGPR r -> row r (+8 hi lanes)
        C[(size_t)row * ldc + colTile * 16 + mn] = acc[r];
    }
}

// ---------------------------------------------------------------- attention dots
__global__ void alpha1_kernel(const float* __restrict__ h1, const float* __restrict__ a1s,
                              const float* __restrict__ a1d,
                              float* __restrict__ as, float* __restrict__ ad) {
    int i = blockIdx.x * blockDim.x + threadIdx.x;
    if (i >= N_NODES * HEADS) return;
    int n = i >> 3, h = i & 7;
    const float* hp = h1 + (size_t)n * D1 + h * HID;
    const float* sp = a1s + h * HID;
    const float* dp = a1d + h * HID;
    float ss = 0.f, dd = 0.f;
#pragma unroll
    for (int c = 0; c < HID; ++c) { float v = hp[c]; ss += v * sp[c]; dd += v * dp[c]; }
    as[i] = ss; ad[i] = dd;
}

__global__ void alpha2_kernel(const float* __restrict__ z, const float* __restrict__ a2s,
                              const float* __restrict__ a2d,
                              float* __restrict__ as, float* __restrict__ ad) {
    int n = blockIdx.x * blockDim.x + threadIdx.x;
    if (n >= N_NODES) return;
    float ss = 0.f, dd = 0.f;
#pragma unroll
    for (int c = 0; c < NCLS; ++c) {
        float v = z[(size_t)n * NCLS_PAD + c];
        ss += v * a2s[c]; dd += v * a2d[c];
    }
    as[n] = ss; ad[n] = dd;
}

// ---------------------------------------------------------------- layer-1 edge phase
__global__ void edge_max1(const long long* __restrict__ ei, const float* __restrict__ as,
                          const float* __restrict__ ad, unsigned* __restrict__ m) {
    int i = blockIdx.x * blockDim.x + threadIdx.x;
    if (i >= E_TOT * HEADS) return;
    int e = i >> 3, h = i & 7;
    int s, d; edge_nodes(ei, e, s, d);
    float v = as[s * HEADS + h] + ad[d * HEADS + h];
    v = fmaxf(v, NEG_SLOPE * v);                 // leaky relu
    atomicMax(&m[d * HEADS + h], encf(v));
}

__global__ void edge_sum1(const long long* __restrict__ ei, const float* __restrict__ as,
                          const float* __restrict__ ad, const unsigned* __restrict__ m,
                          float* __restrict__ den) {
    int i = blockIdx.x * blockDim.x + threadIdx.x;
    if (i >= E_TOT * HEADS) return;
    int e = i >> 3, h = i & 7;
    int s, d; edge_nodes(ei, e, s, d);
    float v = as[s * HEADS + h] + ad[d * HEADS + h];
    v = fmaxf(v, NEG_SLOPE * v);
    atomicAdd(&den[d * HEADS + h], __expf(v - decf(m[d * HEADS + h])));
}

// one block (256 threads) per edge: thread t -> head t>>5, channel t&31
__global__ void aggregate1(const long long* __restrict__ ei, const float* __restrict__ as,
                           const float* __restrict__ ad, const unsigned* __restrict__ m,
                           const float* __restrict__ den, const float* __restrict__ h1,
                           float* __restrict__ agg) {
    int e = blockIdx.x;
    int t = threadIdx.x;
    int h = t >> 5;
    int s, d; edge_nodes(ei, e, s, d);
    float v = as[s * HEADS + h] + ad[d * HEADS + h];
    v = fmaxf(v, NEG_SLOPE * v);
    float alpha = __expf(v - decf(m[d * HEADS + h])) / den[d * HEADS + h];
    atomicAdd(&agg[(size_t)d * D1 + t], alpha * h1[(size_t)s * D1 + t]);
}

// ---------------------------------------------------------------- bias + ELU -> bf16
__global__ void elu_bias_kernel(const float* __restrict__ agg, const float* __restrict__ b1,
                                bf16_t* __restrict__ helu) {
    int i = blockIdx.x * blockDim.x + threadIdx.x;
    if (i >= N_NODES * D1) return;
    int c = i & (D1 - 1);
    float y = agg[i] + b1[c];
    y = (y > 0.f) ? y : (__expf(y) - 1.f);       // ELU(alpha=1)
    helu[i] = (bf16_t)y;
}

// ---------------------------------------------------------------- layer-2 edge phase
__global__ void edge_max2(const long long* __restrict__ ei, const float* __restrict__ as,
                          const float* __restrict__ ad, unsigned* __restrict__ m) {
    int e = blockIdx.x * blockDim.x + threadIdx.x;
    if (e >= E_TOT) return;
    int s, d; edge_nodes(ei, e, s, d);
    float v = as[s] + ad[d];
    v = fmaxf(v, NEG_SLOPE * v);
    atomicMax(&m[d], encf(v));
}

__global__ void edge_sum2(const long long* __restrict__ ei, const float* __restrict__ as,
                          const float* __restrict__ ad, const unsigned* __restrict__ m,
                          float* __restrict__ den) {
    int e = blockIdx.x * blockDim.x + threadIdx.x;
    if (e >= E_TOT) return;
    int s, d; edge_nodes(ei, e, s, d);
    float v = as[s] + ad[d];
    v = fmaxf(v, NEG_SLOPE * v);
    atomicAdd(&den[d], __expf(v - decf(m[d])));
}

__global__ void aggregate2(const long long* __restrict__ ei, const float* __restrict__ as,
                           const float* __restrict__ ad, const unsigned* __restrict__ m,
                           const float* __restrict__ den, const float* __restrict__ z,
                           float* __restrict__ agg) {
    int idx = blockIdx.x * blockDim.x + threadIdx.x;
    if (idx >= E_TOT * NCLS) return;
    int e = idx / NCLS, c = idx - e * NCLS;
    int s, d; edge_nodes(ei, e, s, d);
    float v = as[s] + ad[d];
    v = fmaxf(v, NEG_SLOPE * v);
    float alpha = __expf(v - decf(m[d])) / den[d];
    atomicAdd(&agg[(size_t)d * NCLS + c], alpha * z[(size_t)s * NCLS_PAD + c]);
}

// ---------------------------------------------------------------- bias + log_softmax
__global__ void logsoftmax_kernel(const float* __restrict__ agg, const float* __restrict__ b2,
                                  float* __restrict__ out) {
    int n = blockIdx.x * blockDim.x + threadIdx.x;
    if (n >= N_NODES) return;
    float v[NCLS];
    float mx = -3.4e38f;
#pragma unroll
    for (int c = 0; c < NCLS; ++c) {
        v[c] = agg[(size_t)n * NCLS + c] + b2[c];
        mx = fmaxf(mx, v[c]);
    }
    float sum = 0.f;
#pragma unroll
    for (int c = 0; c < NCLS; ++c) sum += __expf(v[c] - mx);
    float ls = __logf(sum);
#pragma unroll
    for (int c = 0; c < NCLS; ++c) out[(size_t)n * NCLS + c] = v[c] - mx - ls;
}

// ================================================================ launch
static inline int cdiv(long long a, long long b) { return (int)((a + b - 1) / b); }

extern "C" void kernel_launch(void* const* d_in, const int* in_sizes, int n_in,
                              void* d_out, int out_size, void* d_ws, size_t ws_size,
                              hipStream_t stream) {
    const float*     x     = (const float*)d_in[0];
    const long long* ei    = (const long long*)d_in[1];   // int64 edge_index [2,E]
    const float*     W1    = (const float*)d_in[2];
    const float*     a1s   = (const float*)d_in[3];
    const float*     a1d   = (const float*)d_in[4];
    const float*     b1    = (const float*)d_in[5];
    const float*     W2    = (const float*)d_in[6];
    const float*     a2s   = (const float*)d_in[7];
    const float*     a2d   = (const float*)d_in[8];
    const float*     b2    = (const float*)d_in[9];
    float* out = (float*)d_out;

    // ---- workspace layout (256B aligned) ----
    char* p = (char*)d_ws;
    size_t off = 0;
    auto alloc = [&](size_t bytes) { void* r = p + off; off = (off + bytes + 255) & ~(size_t)255; return r; };
    bf16_t*   x16   = (bf16_t*)  alloc((size_t)N_NODES * F_IN * 2);
    bf16_t*   W1t   = (bf16_t*)  alloc((size_t)D1 * F_IN * 2);
    float*    h1    = (float*)   alloc((size_t)N_NODES * D1 * 4);
    float*    as1   = (float*)   alloc((size_t)N_NODES * HEADS * 4);
    float*    ad1   = (float*)   alloc((size_t)N_NODES * HEADS * 4);
    unsigned* m1    = (unsigned*)alloc((size_t)N_NODES * HEADS * 4);
    float*    den1  = (float*)   alloc((size_t)N_NODES * HEADS * 4);
    float*    agg1  = (float*)   alloc((size_t)N_NODES * D1 * 4);
    bf16_t*   helu  = (bf16_t*)  alloc((size_t)N_NODES * D1 * 2);
    bf16_t*   W2t   = (bf16_t*)  alloc((size_t)NCLS_PAD * D1 * 2);
    float*    z2    = (float*)   alloc((size_t)N_NODES * NCLS_PAD * 4);
    float*    as2   = (float*)   alloc((size_t)N_NODES * 4);
    float*    ad2   = (float*)   alloc((size_t)N_NODES * 4);
    unsigned* m2    = (unsigned*)alloc((size_t)N_NODES * 4);
    float*    den2  = (float*)   alloc((size_t)N_NODES * 4);
    float*    agg2  = (float*)   alloc((size_t)N_NODES * NCLS * 4);
    (void)ws_size; (void)in_sizes; (void)n_in; (void)out_size;

    const int B = 256;

    // ---- init accumulators ----
    fill_u32<<<cdiv(N_NODES * HEADS, B), B, 0, stream>>>(m1, ENC_NEG_INF, N_NODES * HEADS);
    fill_u32<<<cdiv(N_NODES * HEADS, B), B, 0, stream>>>((unsigned*)den1, 0u, N_NODES * HEADS);
    fill_u32<<<cdiv(N_NODES * D1, B), B, 0, stream>>>((unsigned*)agg1, 0u, N_NODES * D1);
    fill_u32<<<cdiv(N_NODES, B), B, 0, stream>>>(m2, ENC_NEG_INF, N_NODES);
    fill_u32<<<cdiv(N_NODES, B), B, 0, stream>>>((unsigned*)den2, 0u, N_NODES);
    fill_u32<<<cdiv(N_NODES * NCLS, B), B, 0, stream>>>((unsigned*)agg2, 0u, N_NODES * NCLS);

    // ---- convert operands to bf16 / transposed bf16 ----
    f2bf_kernel<<<cdiv((long long)N_NODES * F_IN, B), B, 0, stream>>>(x, x16, N_NODES * F_IN);
    transpose_bf_kernel<<<cdiv(F_IN * D1, B), B, 0, stream>>>(W1, W1t, F_IN, D1, D1);
    transpose_bf_kernel<<<cdiv(D1 * NCLS_PAD, B), B, 0, stream>>>(W2, W2t, D1, NCLS, NCLS_PAD);

    // ---- layer 1: h1 = x @ W1 (WMMA bf16) ----
    {
        int tiles = (N_NODES / 16) * (D1 / 16);       // 1250 * 16 = 20000 waves
        gemm_bf16_wmma<<<cdiv((long long)tiles * 32, B), B, 0, stream>>>(
            x16, W1t, h1, F_IN, D1 / 16, D1, tiles);
    }
    alpha1_kernel<<<cdiv(N_NODES * HEADS, B), B, 0, stream>>>(h1, a1s, a1d, as1, ad1);
    edge_max1<<<cdiv((long long)E_TOT * HEADS, B), B, 0, stream>>>(ei, as1, ad1, m1);
    edge_sum1<<<cdiv((long long)E_TOT * HEADS, B), B, 0, stream>>>(ei, as1, ad1, m1, den1);
    aggregate1<<<E_TOT, D1, 0, stream>>>(ei, as1, ad1, m1, den1, h1, agg1);
    elu_bias_kernel<<<cdiv((long long)N_NODES * D1, B), B, 0, stream>>>(agg1, b1, helu);

    // ---- layer 2: z2 = elu(h) @ W2 (WMMA bf16, padded to 48 cols) ----
    {
        int tiles = (N_NODES / 16) * (NCLS_PAD / 16); // 1250 * 3 = 3750 waves
        gemm_bf16_wmma<<<cdiv((long long)tiles * 32, B), B, 0, stream>>>(
            helu, W2t, z2, D1, NCLS_PAD / 16, NCLS_PAD, tiles);
    }
    alpha2_kernel<<<cdiv(N_NODES, B), B, 0, stream>>>(z2, a2s, a2d, as2, ad2);
    edge_max2<<<cdiv(E_TOT, B), B, 0, stream>>>(ei, as2, ad2, m2);
    edge_sum2<<<cdiv(E_TOT, B), B, 0, stream>>>(ei, as2, ad2, m2, den2);
    aggregate2<<<cdiv((long long)E_TOT * NCLS, B), B, 0, stream>>>(ei, as2, ad2, m2, den2, z2, agg2);

    // ---- log_softmax -> d_out ----
    logsoftmax_kernel<<<cdiv(N_NODES, B), B, 0, stream>>>(agg2, b2, out);
}
